// Model_48558900248831
// MI455X (gfx1250) — matrix operand order repaired and checked by reading, not compiled, not run on hardware
//
#include <hip/hip_runtime.h>

// CDNA5 / gfx1250. Hetero GraphSAGE: f32 atomic scatter-mean aggregation
// (memory-bound part) + bf16 WMMA (f32 accumulate) for all row-wise linears.
// Weights are pre-swizzled into per-lane WMMA fragment order so each lane's
// B fragment is one contiguous 32-byte (v16bf) load.

typedef __attribute__((ext_vector_type(16))) __bf16 v16bf;
typedef __attribute__((ext_vector_type(8)))  float  v8f;

#define HDIM 64

__global__ __launch_bounds__(256)
void k_zero(float* __restrict__ p, long long n) {
  long long i = (long long)blockIdx.x * blockDim.x + threadIdx.x;
  if (i < n) p[i] = 0.0f;
}

// x_movie = movie_x @ w_movie + b_movie + movie_emb   (F=20, one thread per (m,h))
__global__ __launch_bounds__(256)
void k_movie_proj(const float* __restrict__ movie_x, const float* __restrict__ w_movie,
                  const float* __restrict__ b_movie, const float* __restrict__ movie_emb,
                  float* __restrict__ xm, int M) {
  long long t = (long long)blockIdx.x * blockDim.x + threadIdx.x;
  int m = (int)(t >> 6), h = (int)t & 63;
  if (m >= M) return;
  float acc = b_movie[h] + movie_emb[(size_t)m * HDIM + h];
  const float* row = movie_x + (size_t)m * 20;
#pragma unroll
  for (int f = 0; f < 20; ++f) acc += row[f] * w_movie[f * HDIM + h];
  xm[(size_t)m * HDIM + h] = acc;
}

__global__ __launch_bounds__(256)
void k_degrees(const int* __restrict__ src, const int* __restrict__ dst,
               float* __restrict__ degu, float* __restrict__ degm, int E) {
  int e = blockIdx.x * blockDim.x + threadIdx.x;
  if (e >= E) return;
  atomicAdd(&degu[src[e]], 1.0f);
  atomicAdd(&degm[dst[e]], 1.0f);
}

// both-direction edge scatter: aggu[src] += xm[dst], aggm[dst] += xu[src]
// 16 threads per edge, each owning a float4 chunk of the 64-wide feature.
__global__ __launch_bounds__(256)
void k_scatter(const int* __restrict__ src, const int* __restrict__ dst,
               const float* __restrict__ xu, const float* __restrict__ xm,
               float* __restrict__ aggu, float* __restrict__ aggm, int E) {
  long long t = (long long)blockIdx.x * blockDim.x + threadIdx.x;
  int e = (int)(t >> 4);
  if (e >= E) return;
  int q = ((int)t & 15) << 2;
  int s = src[e], d = dst[e];
  float4 vm = *(const float4*)(xm + (size_t)d * HDIM + q);
  float4 vu = *(const float4*)(xu + (size_t)s * HDIM + q);
  float* au = aggu + (size_t)s * HDIM + q;
  float* am = aggm + (size_t)d * HDIM + q;
  atomicAdd(au + 0, vm.x); atomicAdd(au + 1, vm.y);
  atomicAdd(au + 2, vm.z); atomicAdd(au + 3, vm.w);
  atomicAdd(am + 0, vu.x); atomicAdd(am + 1, vu.y);
  atomicAdd(am + 2, vu.z); atomicAdd(am + 3, vu.w);
}

// Pack [Wl;Wr] (each 64x64 f32 row-major, combined K=128) into bf16 in
// per-lane WMMA B-fragment order:
//   flat index = ((((ks*4 + t)*2 + hk)*16 + lm)*16 + j)
//   fragment element j:  k = ks*32 + (j&7) + ((j&8)<<1) + (hk<<3)
//   column n = t*16 + lm
// so each lane's 16-element fragment is 32 contiguous bytes.
__global__ __launch_bounds__(256)
void k_pack_w(const float* __restrict__ Wl, const float* __restrict__ Wr,
              __bf16* __restrict__ wpk) {
  int i = blockIdx.x * blockDim.x + threadIdx.x;   // 0..8191
  if (i >= 128 * HDIM) return;
  int j  = i & 15;
  int lm = (i >> 4) & 15;
  int hk = (i >> 8) & 1;
  int t  = (i >> 9) & 3;
  int ks = (i >> 11) & 3;
  int k = ks * 32 + (j & 7) + ((j & 8) << 1) + (hk << 3);
  int n = t * 16 + lm;
  float v = (k < HDIM) ? Wl[k * HDIM + n] : Wr[(k - HDIM) * HDIM + n];
  wpk[i] = (__bf16)v;
}

// y = relu?( (agg/deg) @ Wl + x @ Wr + b )  as one K=128 bf16 WMMA GEMM.
// 4 waves/block, one 16-row tile per wave, 4 N-tiles x 4 K-steps of
// v_wmma_f32_16x16x32_bf16. B fragments come straight from the swizzled
// global weight block (L2/WGP$-resident, coalesced 32B per lane).
__global__ __launch_bounds__(128)
void k_sage_wmma(const float* __restrict__ agg, const float* __restrict__ deg,
                 const float* __restrict__ xroot, const __bf16* __restrict__ wpk,
                 const float* __restrict__ bias, float* __restrict__ y,
                 int nrows, int do_relu) {
  const int wv = threadIdx.x >> 5;
  const int ln = threadIdx.x & 31;
  const int hk = ln >> 4;              // half-wave (K split per ISA layout)
  const int lm = ln & 15;
  const int m0 = (blockIdx.x * 4 + wv) * 16;
  if (m0 >= nrows) return;

  const int m = m0 + lm;               // A-matrix row owned by this lane
  const float inv = 1.0f / fmaxf(deg[m], 1.0f);
  const float* arow = agg + (size_t)m * HDIM;
  const float* xrow = xroot + (size_t)m * HDIM;
  const v16bf* wfr = (const v16bf*)wpk;

  v8f acc[4];
#pragma unroll
  for (int t = 0; t < 4; ++t) {
    float bv = bias[t * 16 + lm];      // C/D: col = lane%16 (+16t)
#pragma unroll
    for (int r = 0; r < 8; ++r) acc[t][r] = bv;
  }

#pragma unroll
  for (int ks = 0; ks < 4; ++ks) {     // K = 128 in 4 steps of 32
    // A fragment: 16-bit A layout — VGPR i holds K pair; i<4 -> K base 0,
    // i>=4 -> +16; +8 for lanes 16..31. Pairs merge into b128 loads.
    v16bf a;
#pragma unroll
    for (int i = 0; i < 8; ++i) {
      int kk = ks * 32 + ((i & 4) << 2) + ((i & 3) << 1) + (hk << 3);
      float f0, f1;
      if (kk < HDIM) { f0 = arow[kk] * inv;      f1 = arow[kk + 1] * inv; }
      else           { f0 = xrow[kk - HDIM];     f1 = xrow[kk - HDIM + 1]; }
      a[2 * i]     = (__bf16)f0;
      a[2 * i + 1] = (__bf16)f1;
    }
    // B fragments: one contiguous v16bf per lane from the swizzled block.
#pragma unroll
    for (int t = 0; t < 4; ++t) {
      v16bf b = wfr[(((ks << 2) + t) * 2 + hk) * 16 + lm];
      acc[t] = __builtin_amdgcn_wmma_f32_16x16x32_bf16(
          false, a, false, b, (short)0, acc[t], false, false);
    }
  }

#pragma unroll
  for (int t = 0; t < 4; ++t) {
    const int col = t * 16 + lm;
#pragma unroll
    for (int r = 0; r < 8; ++r) {      // D: VGPR r -> row r + 8*half
      float v = acc[t][r];
      if (do_relu) v = fmaxf(v, 0.0f);
      y[(size_t)(m0 + r + 8 * hk) * HDIM + col] = v;
    }
  }
}

// out[e][c] = concat(u2[els[e]], m2[eld[e]]) @ cls_W + cls_b   (C=7)
__global__ __launch_bounds__(256)
void k_classify(const int* __restrict__ els, const int* __restrict__ eld,
                const float* __restrict__ u2, const float* __restrict__ m2,
                const float* __restrict__ clsW, const float* __restrict__ clsb,
                float* __restrict__ out, int EL) {
  long long t = (long long)blockIdx.x * blockDim.x + threadIdx.x;
  int e = (int)(t >> 3);
  int c = (int)t & 7;
  if (e >= EL || c >= 7) return;
  const float* ur = u2 + (size_t)els[e] * HDIM;
  const float* mr = m2 + (size_t)eld[e] * HDIM;
  float acc = clsb[c];
#pragma unroll 8
  for (int k = 0; k < HDIM; ++k) acc += ur[k] * clsW[k * 7 + c];
#pragma unroll 8
  for (int k = 0; k < HDIM; ++k) acc += mr[k] * clsW[(HDIM + k) * 7 + c];
  out[(size_t)e * 7 + c] = acc;
}

static inline int cdiv_i(long long a, long long b) { return (int)((a + b - 1) / b); }

extern "C" void kernel_launch(void* const* d_in, const int* in_sizes, int n_in,
                              void* d_out, int out_size, void* d_ws, size_t ws_size,
                              hipStream_t stream) {
  const int U  = in_sizes[0];
  const int M  = in_sizes[1];
  const int E  = in_sizes[3];
  const int EL = in_sizes[5];

  const float* movie_x   = (const float*)d_in[2];
  const int*   edge_src  = (const int*)d_in[3];
  const int*   edge_dst  = (const int*)d_in[4];
  const int*   el_src    = (const int*)d_in[5];
  const int*   el_dst    = (const int*)d_in[6];
  const float* user_emb  = (const float*)d_in[7];   // user_node_id = arange -> identity gather
  const float* movie_emb = (const float*)d_in[8];
  const float* w_movie   = (const float*)d_in[9];
  const float* b_movie   = (const float*)d_in[10];
  const float* l1_rates_Wl = (const float*)d_in[11];
  const float* l1_rates_bl = (const float*)d_in[12];
  const float* l1_rates_Wr = (const float*)d_in[13];
  const float* l1_rev_Wl   = (const float*)d_in[14];
  const float* l1_rev_bl   = (const float*)d_in[15];
  const float* l1_rev_Wr   = (const float*)d_in[16];
  const float* l2_rates_Wl = (const float*)d_in[17];
  const float* l2_rates_bl = (const float*)d_in[18];
  const float* l2_rates_Wr = (const float*)d_in[19];
  const float* l2_rev_Wl   = (const float*)d_in[20];
  const float* l2_rev_bl   = (const float*)d_in[21];
  const float* l2_rev_Wr   = (const float*)d_in[22];
  const float* cls_W = (const float*)d_in[23];
  const float* cls_b = (const float*)d_in[24];

  // workspace layout (f32): [deg_u][deg_m][agg_u][agg_m][xm][u1][m1][u2][m2] + bf16 weights
  float* ws = (float*)d_ws;
  size_t off = 0;
  float* deg_u = ws + off; off += (size_t)U;
  float* deg_m = ws + off; off += (size_t)M;
  float* agg_u = ws + off; off += (size_t)U * HDIM;
  float* agg_m = ws + off; off += (size_t)M * HDIM;
  float* xm    = ws + off; off += (size_t)M * HDIM;
  float* u1    = ws + off; off += (size_t)U * HDIM;
  float* m1    = ws + off; off += (size_t)M * HDIM;
  float* u2    = ws + off; off += (size_t)U * HDIM;
  float* m2    = ws + off; off += (size_t)M * HDIM;
  __bf16* wc_l1rev   = (__bf16*)(ws + off);
  __bf16* wc_l1rates = wc_l1rev + 128 * HDIM;
  __bf16* wc_l2rev   = wc_l1rates + 128 * HDIM;
  __bf16* wc_l2rates = wc_l2rev + 128 * HDIM;

  const int BT = 256;
  const long long nzero_all = (long long)(U + M) + (long long)(U + M) * HDIM;
  const long long nzero_agg = (long long)(U + M) * HDIM;

  // ---- setup ----
  k_zero<<<cdiv_i(nzero_all, BT), BT, 0, stream>>>(deg_u, nzero_all);
  k_movie_proj<<<cdiv_i((long long)M * HDIM, BT), BT, 0, stream>>>(
      movie_x, w_movie, b_movie, movie_emb, xm, M);
  k_degrees<<<cdiv_i(E, BT), BT, 0, stream>>>(edge_src, edge_dst, deg_u, deg_m, E);
  k_pack_w<<<cdiv_i(128 * HDIM, BT), BT, 0, stream>>>(l1_rev_Wl, l1_rev_Wr, wc_l1rev);
  k_pack_w<<<cdiv_i(128 * HDIM, BT), BT, 0, stream>>>(l1_rates_Wl, l1_rates_Wr, wc_l1rates);
  k_pack_w<<<cdiv_i(128 * HDIM, BT), BT, 0, stream>>>(l2_rev_Wl, l2_rev_Wr, wc_l2rev);
  k_pack_w<<<cdiv_i(128 * HDIM, BT), BT, 0, stream>>>(l2_rates_Wl, l2_rates_Wr, wc_l2rates);

  const int gridU = cdiv_i(cdiv_i(U, 16), 4);
  const int gridM = cdiv_i(cdiv_i(M, 16), 4);
  const int gridE = cdiv_i((long long)E * 16, BT);

  // ---- layer 1 ----
  k_scatter<<<gridE, BT, 0, stream>>>(edge_src, edge_dst, user_emb, xm,
                                      agg_u, agg_m, E);
  k_sage_wmma<<<gridU, 128, 0, stream>>>(agg_u, deg_u, user_emb, wc_l1rev,
                                         l1_rev_bl, u1, U, 1);
  k_sage_wmma<<<gridM, 128, 0, stream>>>(agg_m, deg_m, xm, wc_l1rates,
                                         l1_rates_bl, m1, M, 1);

  // ---- layer 2 ----
  k_zero<<<cdiv_i(nzero_agg, BT), BT, 0, stream>>>(agg_u, nzero_agg);
  k_scatter<<<gridE, BT, 0, stream>>>(edge_src, edge_dst, u1, m1,
                                      agg_u, agg_m, E);
  k_sage_wmma<<<gridU, 128, 0, stream>>>(agg_u, deg_u, u1, wc_l2rev,
                                         l2_rev_bl, u2, U, 0);
  k_sage_wmma<<<gridM, 128, 0, stream>>>(agg_m, deg_m, m1, wc_l2rates,
                                         l2_rates_bl, m2, M, 0);

  // ---- classifier ----
  k_classify<<<cdiv_i((long long)EL * 8, BT), BT, 0, stream>>>(
      el_src, el_dst, u2, m2, cls_W, cls_b, (float*)d_out, EL);
}
